// TransformerHybridActor_68642167324674
// MI455X (gfx1250) — compile-verified, matrix operand order checked by reading, not run on hardware
//
#include <hip/hip_runtime.h>

// ---------------------------------------------------------------------------
// GATv2 hybrid actor forward for MI455X (gfx1250, wave32, WMMA).
// Dense projections + scoring einsum use v_wmma_f32_16x16x32_f16.
// All WMMA fragment loads are branch-free (address clamp + value select) so
// they lower to vectorized global_load_b64 + v_cndmask, not exec-mask clauses.
// Edge softmax/aggregation uses f32 global atomics with float4 gathers.
// ---------------------------------------------------------------------------

typedef __attribute__((ext_vector_type(16))) _Float16 v16h;
typedef __attribute__((ext_vector_type(8)))  float    v8f;

static constexpr int Bc = 128, Tc = 512, Vc = 32, Rc = 8;
static constexpr int NT = Bc * Tc;      // 65536 task nodes
static constexpr int NV = Bc * Vc;      // 4096 vehicle nodes
static constexpr int NR = Bc * Rc;      // 1024 rsu nodes
static constexpr int E_DAG = Bc * 1024; // 131072
static constexpr int E_V2V = Bc * 256;  // 32768
static constexpr int E_VI  = Bc * Vc * Rc; // 32768
static constexpr int KTGT = Rc + Vc;    // 40 offload targets
static constexpr float NEGV = -1e9f;

// K index of packed pair v (0..7) for the 16-bit 16x32 WMMA A/B fragment.
__device__ __forceinline__ int wmma_kpair(int v, int half) {
  return (v < 4) ? (half * 8 + 2 * v) : (16 + half * 8 + 2 * (v - 4));
}

__device__ __forceinline__ void atomicMaxFloat(float* addr, float val) {
  unsigned int* ua = (unsigned int*)addr;
  unsigned int old = *ua;
  while (true) {
    float f = __uint_as_float(old);
    if (f >= val) break;
    unsigned int assumed = old;
    old = atomicCAS(ua, assumed, __float_as_uint(val));
    if (old == assumed) break;
  }
}

// ---------------------------------------------------------------------------
__global__ void fill_f32(float* p, float v, int n) {
  int t = blockIdx.x * blockDim.x + threadIdx.x;
  if (t < n) p[t] = v;
}

// ---------------------------------------------------------------------------
// C[M,N] = A[M,K] @ W[K,N] + bias.  fp32 in/out, fp16 WMMA inner product.
// One wave computes a 16x(16*NTILES) strip: A fragment loaded once per k-step,
// feeds NTILES v_wmma issues. 8 waves / block (one m-tile each).
// Requires: M % 16 == 0, N % (16*NTILES) == 0, K even (tail zero-padded).
template <int NTILES>
__global__ void gemm_strip_wmma(const float* __restrict__ A,
                                const float* __restrict__ W,
                                const float* __restrict__ bias,
                                float* __restrict__ C,
                                int M, int K, int N) {
  int lane = threadIdx.x & 31;
  int wave = threadIdx.x >> 5;
  int half = lane >> 4;
  int mn = lane & 15;
  int m0 = (blockIdx.y * 8 + wave) * 16;
  if (m0 >= M) return; // wave-uniform
  int nbase = blockIdx.x * (16 * NTILES);
  int arow = m0 + mn;
  v8f zero = {};
  v8f acc[NTILES];
#pragma unroll
  for (int i = 0; i < NTILES; ++i) acc[i] = zero;
  const float* Arow = A + (size_t)arow * K;
  for (int k0 = 0; k0 < K; k0 += 32) {
    v16h af;
#pragma unroll
    for (int v = 0; v < 8; ++v) {
      int k = k0 + wmma_kpair(v, half);
      int kc = (k + 1 < K) ? k : (K - 2); // clamp; zero-select below
      float2 av = *(const float2*)(Arow + kc);
      af[2 * v]     = (_Float16)((k < K) ? av.x : 0.f);
      af[2 * v + 1] = (_Float16)((k + 1 < K) ? av.y : 0.f);
    }
#pragma unroll
    for (int nt = 0; nt < NTILES; ++nt) {
      int bcol = nbase + nt * 16 + mn;
      v16h bf;
#pragma unroll
      for (int v = 0; v < 8; ++v) {
        int k = k0 + wmma_kpair(v, half);
        int kc = (k + 1 < K) ? k : (K - 2);
        float b0 = W[(size_t)kc * N + bcol];
        float b1 = W[(size_t)(kc + 1) * N + bcol];
        bf[2 * v]     = (_Float16)((k < K) ? b0 : 0.f);
        bf[2 * v + 1] = (_Float16)((k + 1 < K) ? b1 : 0.f);
      }
      acc[nt] = __builtin_amdgcn_wmma_f32_16x16x32_f16(
          false, af, false, bf, (short)0, acc[nt], false, false);
    }
  }
#pragma unroll
  for (int nt = 0; nt < NTILES; ++nt) {
#pragma unroll
    for (int r = 0; r < 8; ++r) {
      int m = m0 + r + 8 * half;
      int n = nbase + nt * 16 + mn;
      C[(size_t)m * N + n] = acc[nt][r] + bias[n];
    }
  }
}

static inline int cdiv(int a, int b) { return (a + b - 1) / b; }

static void launch_gemm(hipStream_t st, const float* A, const float* W,
                        const float* bias, float* C, int M, int K, int N) {
  dim3 blk(256);
  if (N % 128 == 0) {
    gemm_strip_wmma<8><<<dim3(N / 128, cdiv(M, 128)), blk, 0, st>>>(A, W, bias, C, M, K, N);
  } else { // N % 64 == 0
    gemm_strip_wmma<4><<<dim3(N / 64, cdiv(M, 128)), blk, 0, st>>>(A, W, bias, C, M, K, N);
  }
}

// ---------------------------------------------------------------------------
// ge[e, :] = edge_attr[e, 0:2] @ W_e[2, HD]
__global__ void edge_proj_k(const float* __restrict__ ea,
                            const float* __restrict__ We,
                            float* __restrict__ ge, int E, int HD) {
  int t = blockIdx.x * blockDim.x + threadIdx.x;
  if (t >= E * HD) return;
  int e = t / HD, j = t % HD;
  float2 a = *(const float2*)(ea + e * 2);
  ge[t] = a.x * We[j] + a.y * We[HD + j];
}

// column mean of [E,2] edge attrs (fill_value='mean' for self loops)
__global__ void col_mean2_k(const float* __restrict__ ea, float* out, int E) {
  int c = threadIdx.x;
  if (c < 2) {
    float s = 0.f;
    for (int e = 0; e < E; ++e) s += ea[e * 2 + c];
    out[c] = s / (float)E;
  }
}

__global__ void ge_fill_k(const float* __restrict__ mean2,
                          const float* __restrict__ We,
                          float* __restrict__ gefill, int HD) {
  int j = blockIdx.x * blockDim.x + threadIdx.x;
  if (j < HD) gefill[j] = mean2[0] * We[j] + mean2[1] * We[HD + j];
}

// ---------------------------------------------------------------------------
// logit[e,h] = att_h . leaky_relu(gl[src] + gr[dst] (+ ge)); segment max -> m
// float4 gathers (Dout % 4 == 0; all row bases 16B aligned).
__global__ void edge_logits_k(const float* __restrict__ gl,
                              const float* __restrict__ gr,
                              const float* __restrict__ ge,
                              const float* __restrict__ gefill,
                              const int* __restrict__ src,
                              const int* __restrict__ dst,
                              const float* __restrict__ att,
                              float* __restrict__ logit,
                              float* __restrict__ mbuf,
                              int E0, int Etot, int H, int Dout) {
  int t = blockIdx.x * blockDim.x + threadIdx.x;
  if (t >= Etot * H) return;
  int e = t / H, h = t % H;
  int s = (e < E0) ? src[e] : (e - E0);
  int d = (e < E0) ? dst[e] : (e - E0);
  int HD = H * Dout;
  const float* glp = gl + (size_t)s * HD + h * Dout;
  const float* grp = gr + (size_t)d * HD + h * Dout;
  const float* gep = nullptr;
  if (ge) gep = (e < E0) ? (ge + (size_t)e * HD + h * Dout) : (gefill + h * Dout);
  const float* ap = att + h * Dout;
  float acc = 0.f;
  for (int j = 0; j < Dout; j += 4) {
    float4 a = *(const float4*)(glp + j);
    float4 b = *(const float4*)(grp + j);
    float4 w = *(const float4*)(ap + j);
    float4 c = {0.f, 0.f, 0.f, 0.f};
    if (gep) c = *(const float4*)(gep + j);
    float v0 = a.x + b.x + c.x; v0 = (v0 > 0.f) ? v0 : 0.2f * v0;
    float v1 = a.y + b.y + c.y; v1 = (v1 > 0.f) ? v1 : 0.2f * v1;
    float v2 = a.z + b.z + c.z; v2 = (v2 > 0.f) ? v2 : 0.2f * v2;
    float v3 = a.w + b.w + c.w; v3 = (v3 > 0.f) ? v3 : 0.2f * v3;
    acc += v0 * w.x + v1 * w.y + v2 * w.z + v3 * w.w;
  }
  logit[t] = acc;
  atomicMaxFloat(&mbuf[d * H + h], acc);
}

// ex = exp(logit - m[dst]); den[dst] += ex
__global__ void edge_exp_k(const float* __restrict__ logit,
                           const float* __restrict__ mbuf,
                           const int* __restrict__ dst,
                           float* __restrict__ exb,
                           float* __restrict__ den,
                           int E0, int Etot, int H) {
  int t = blockIdx.x * blockDim.x + threadIdx.x;
  if (t >= Etot * H) return;
  int e = t / H, h = t % H;
  int d = (e < E0) ? dst[e] : (e - E0);
  float ex = __expf(logit[t] - mbuf[d * H + h]);
  exb[t] = ex;
  atomicAdd(&den[d * H + h], ex);
}

// agg[dst,h,:] += (ex/den) * gl[src,h,:]   (4 elements / thread)
__global__ void edge_scatter_k(const float* __restrict__ gl,
                               const float* __restrict__ exb,
                               const float* __restrict__ den,
                               const int* __restrict__ src,
                               const int* __restrict__ dst,
                               float* __restrict__ agg,
                               int E0, int Etot, int H, int Dout) {
  int HD = H * Dout;
  int HD4 = HD >> 2;
  int t = blockIdx.x * blockDim.x + threadIdx.x;
  if (t >= Etot * HD4) return;
  int e = t / HD4, r = (t % HD4) << 2;
  int h = r / Dout; // r..r+3 stay within one head (Dout % 4 == 0)
  int s = (e < E0) ? src[e] : (e - E0);
  int d = (e < E0) ? dst[e] : (e - E0);
  float dv = den[d * H + h];
  float alpha = exb[e * H + h] / ((dv > 0.f) ? dv : 1.f);
  float4 g = *(const float4*)(gl + (size_t)s * HD + r);
  float* ap = agg + (size_t)d * HD + r;
  atomicAdd(ap + 0, alpha * g.x);
  atomicAdd(ap + 1, alpha * g.y);
  atomicAdd(ap + 2, alpha * g.z);
  atomicAdd(ap + 3, alpha * g.w);
}

// out = concat(agg) + bias   OR   mean over heads + bias
__global__ void gat_finalize_k(const float* __restrict__ agg,
                               const float* __restrict__ bias,
                               float* __restrict__ out,
                               int Nd, int H, int Dout, int concat) {
  int HD = H * Dout;
  int t = blockIdx.x * blockDim.x + threadIdx.x;
  if (concat) {
    if (t >= Nd * HD) return;
    out[t] = agg[t] + bias[t % HD];
  } else {
    if (t >= Nd * Dout) return;
    int n = t / Dout, dd = t % Dout;
    float s = 0.f;
    for (int h = 0; h < H; ++h) s += agg[(size_t)n * HD + h * Dout + dd];
    out[t] = s / (float)H + bias[dd];
  }
}

// ---------------------------------------------------------------------------
__global__ void layernorm_k(const float* __restrict__ x,
                            const float* __restrict__ g,
                            const float* __restrict__ b,
                            float* __restrict__ out, int N, int D, int elu) {
  int n = blockIdx.x * blockDim.x + threadIdx.x;
  if (n >= N) return;
  const float* row = x + (size_t)n * D;
  float mu = 0.f;
  for (int j = 0; j < D; j += 4) {
    float4 v = *(const float4*)(row + j);
    mu += v.x + v.y + v.z + v.w;
  }
  mu /= (float)D;
  float var = 0.f;
  for (int j = 0; j < D; j += 4) {
    float4 v = *(const float4*)(row + j);
    float d0 = v.x - mu, d1 = v.y - mu, d2 = v.z - mu, d3 = v.w - mu;
    var += d0 * d0 + d1 * d1 + d2 * d2 + d3 * d3;
  }
  var /= (float)D;
  float rs = rsqrtf(var + 1e-5f);
  for (int j = 0; j < D; ++j) {
    float v = (row[j] - mu) * rs * g[j] + b[j];
    if (elu) v = (v > 0.f) ? v : expm1f(v);
    out[(size_t)n * D + j] = v;
  }
}

// mode 0: a+b ; mode 1: elu(a (+b))
__global__ void ewise_k(const float* __restrict__ a, const float* __restrict__ b,
                        float* __restrict__ out, int n, int mode) {
  int t = blockIdx.x * blockDim.x + threadIdx.x;
  if (t >= n) return;
  float v = a[t] + (b ? b[t] : 0.f);
  if (mode == 1) v = (v > 0.f) ? v : expm1f(v);
  out[t] = v;
}

// out[b,d] = mean_i x[b*C+i, d]
__global__ void mean_pool_k(const float* __restrict__ x, float* __restrict__ out,
                            int Bn, int C, int D) {
  int t = blockIdx.x * blockDim.x + threadIdx.x;
  if (t >= Bn * D) return;
  int b = t / D, d = t % D;
  float s = 0.f;
  for (int i = 0; i < C; ++i) s += x[((size_t)b * C + i) * D + d];
  out[t] = s / (float)C;
}

// ---------------------------------------------------------------------------
// scores[b,t,k] = node_emb[b,t,:] . target[b,k,:] / 8 ; mask -> -1e9
// target = concat(rsu_emb[b] (8), veh_emb[b] (32)); WMMA K=64 (2 steps).
// Branch-free fragment loads: target row index clamped, value zero-selected.
__global__ void scores_wmma_k(const float* __restrict__ node_emb,
                              const float* __restrict__ veh_emb,
                              const float* __restrict__ rsu_emb,
                              const unsigned char* __restrict__ mask,
                              float* __restrict__ logits) {
  int bb = blockIdx.z;
  int t0 = blockIdx.y * 16;
  int n0 = blockIdx.x * 16;
  int lane = threadIdx.x;
  int half = lane >> 4;
  int mn = lane & 15;
  const float* Arow = node_emb + ((size_t)bb * Tc + t0 + mn) * 64;
  int ncol = n0 + mn;
  int nc = (ncol < KTGT) ? ncol : (KTGT - 1);
  bool nvalid = ncol < KTGT;
  const float* tgt = (nc < Rc)
      ? (rsu_emb + ((size_t)bb * Rc + nc) * 64)
      : (veh_emb + ((size_t)bb * Vc + (nc - Rc)) * 64);
  v8f acc = {};
  for (int k0 = 0; k0 < 64; k0 += 32) {
    v16h af, bf;
#pragma unroll
    for (int v = 0; v < 8; ++v) {
      int k = k0 + wmma_kpair(v, half);
      float2 av = *(const float2*)(Arow + k);
      af[2 * v]     = (_Float16)av.x;
      af[2 * v + 1] = (_Float16)av.y;
      float2 bv = *(const float2*)(tgt + k);
      bf[2 * v]     = (_Float16)(nvalid ? bv.x : 0.f);
      bf[2 * v + 1] = (_Float16)(nvalid ? bv.y : 0.f);
    }
    acc = __builtin_amdgcn_wmma_f32_16x16x32_f16(false, af, false, bf,
                                                 (short)0, acc, false, false);
  }
#pragma unroll
  for (int r = 0; r < 8; ++r) {
    int t = t0 + r + 8 * half;
    if (nvalid) {
      bool mk = mask[(size_t)bb * Tc + t] != 0;
      logits[((size_t)bb * Tc + t) * KTGT + ncol] = mk ? acc[r] * 0.125f : NEGV;
    }
  }
}

// power head: pf=[graph_emb|veh_mean] (128) -> relu(W1)->32 -> sigmoid(W2)->1
__global__ void power_k(const float* __restrict__ gemb,
                        const float* __restrict__ vmean,
                        const float* __restrict__ W1, const float* __restrict__ b1,
                        const float* __restrict__ W2, const float* __restrict__ b2,
                        float* __restrict__ out) {
  int b = blockIdx.x;
  int j = threadIdx.x; // 32 threads
  __shared__ float hsh[32];
  float acc = b1[j];
  for (int i = 0; i < 128; ++i) {
    float pf = (i < 64) ? gemb[b * 64 + i] : vmean[b * 64 + (i - 64)];
    acc += pf * W1[i * 32 + j];
  }
  hsh[j] = fmaxf(acc, 0.f);
  __syncthreads();
  if (j == 0) {
    float s = b2[0];
    for (int k = 0; k < 32; ++k) s += hsh[k] * W2[k];
    out[b] = 1.f / (1.f + __expf(-s));
  }
}

// ---------------------------------------------------------------------------
// Host orchestration
// ---------------------------------------------------------------------------
struct GatP {
  const float *W_l, *b_l, *W_r, *b_r, *att, *bias, *W_e;
};

static void run_gatv2(hipStream_t st,
                      const float* x_src, int Ns, int Kin_src,
                      const float* x_dst, int Nd, int Kin_dst,
                      const int* src, const int* dst, int E0, bool self_loops,
                      const float* edge_attr, const float* eamean,
                      const GatP& P, int H, int Dout, bool concat,
                      float* out,
                      float* gl, float* gr, float* ge, float* gefill,
                      float* logit, float* exb, float* mbuf, float* den,
                      float* agg) {
  const int HD = H * Dout;
  const int Etot = E0 + (self_loops ? Nd : 0);
  launch_gemm(st, x_src, P.W_l, P.b_l, gl, Ns, Kin_src, HD);
  launch_gemm(st, x_dst, P.W_r, P.b_r, gr, Nd, Kin_dst, HD);
  const float* gep = nullptr;
  const float* gfp = nullptr;
  if (edge_attr) {
    edge_proj_k<<<cdiv(E0 * HD, 256), 256, 0, st>>>(edge_attr, P.W_e, ge, E0, HD);
    gep = ge;
    if (self_loops) {
      ge_fill_k<<<cdiv(HD, 128), 128, 0, st>>>(eamean, P.W_e, gefill, HD);
      gfp = gefill;
    }
  }
  fill_f32<<<cdiv(Nd * H, 256), 256, 0, st>>>(mbuf, -1e30f, Nd * H);
  fill_f32<<<cdiv(Nd * H, 256), 256, 0, st>>>(den, 0.f, Nd * H);
  fill_f32<<<cdiv(Nd * HD, 256), 256, 0, st>>>(agg, 0.f, Nd * HD);
  edge_logits_k<<<cdiv(Etot * H, 256), 256, 0, st>>>(
      gl, gr, gep, gfp, src, dst, P.att, logit, mbuf, E0, Etot, H, Dout);
  edge_exp_k<<<cdiv(Etot * H, 256), 256, 0, st>>>(
      logit, mbuf, dst, exb, den, E0, Etot, H);
  edge_scatter_k<<<cdiv(Etot * (HD / 4), 256), 256, 0, st>>>(
      gl, exb, den, src, dst, agg, E0, Etot, H, Dout);
  int fin_n = concat ? Nd * HD : Nd * Dout;
  gat_finalize_k<<<cdiv(fin_n, 256), 256, 0, st>>>(
      agg, P.bias, out, Nd, H, Dout, concat ? 1 : 0);
}

static GatP gatp(void* const* p, int base, bool has_We) {
  GatP g;
  g.W_l = (const float*)p[base + 0];
  g.b_l = (const float*)p[base + 1];
  g.W_r = (const float*)p[base + 2];
  g.b_r = (const float*)p[base + 3];
  g.att = (const float*)p[base + 4];
  g.bias = (const float*)p[base + 5];
  g.W_e = has_We ? (const float*)p[base + 6] : nullptr;
  return g;
}

extern "C" void kernel_launch(void* const* d_in, const int* in_sizes, int n_in,
                              void* d_out, int out_size, void* d_ws, size_t ws_size,
                              hipStream_t stream) {
  (void)in_sizes; (void)n_in; (void)out_size; (void)ws_size;
  hipStream_t st = stream;

  // ---- inputs (setup_inputs insertion order; params flattened depth-first) --
  const float* task_x = (const float*)d_in[0]; // [NT,16]
  const float* veh_x  = (const float*)d_in[1]; // [NV,12]
  const float* rsu_x  = (const float*)d_in[2]; // [NR,8]
  const float* ea_v2v = (const float*)d_in[3]; // [E_V2V,2]
  const float* ea_v2i = (const float*)d_in[4]; // [E_VI,2]
  const float* ea_i2v = (const float*)d_in[5]; // [E_VI,2]
  GatP dag1 = gatp(d_in, 6, false);
  const float* ln1_g = (const float*)d_in[12];
  const float* ln1_b = (const float*)d_in[13];
  GatP dag2 = gatp(d_in, 14, false);
  GatP h1_v2v = gatp(d_in, 20, true);
  GatP h1_v2i = gatp(d_in, 27, true);
  GatP h1_i2v = gatp(d_in, 34, true);
  GatP h2_v2v = gatp(d_in, 41, true);
  GatP h2_v2i = gatp(d_in, 48, true);
  GatP h2_i2v = gatp(d_in, 55, true);
  const float* lnv_g = (const float*)d_in[62];
  const float* lnv_b = (const float*)d_in[63];
  const float* lnr_g = (const float*)d_in[64];
  const float* lnr_b = (const float*)d_in[65];
  const float* pW1 = (const float*)d_in[66];
  const float* pb1 = (const float*)d_in[67];
  const float* pW2 = (const float*)d_in[68];
  const float* pb2 = (const float*)d_in[69];
  const int* dag_ei = (const int*)d_in[70];                   // [2, E_DAG]
  const unsigned char* mask = (const unsigned char*)d_in[71]; // [NT] bool
  const int* v2v_ei = (const int*)d_in[72];                   // [2, E_V2V]
  const int* v2i_ei = (const int*)d_in[73];                   // [2, E_VI]
  const int* i2v_ei = (const int*)d_in[74];                   // [2, E_VI]

  const int* dag_s = dag_ei;           const int* dag_d = dag_ei + E_DAG;
  const int* v2v_s = v2v_ei;           const int* v2v_d = v2v_ei + E_V2V;
  const int* v2i_s = v2i_ei;           const int* v2i_d = v2i_ei + E_VI;
  const int* i2v_s = i2v_ei;           const int* i2v_d = i2v_ei + E_VI;

  // ---- workspace carve-up (floats) ----------------------------------------
  size_t off = 0;
  auto alloc = [&](size_t nf) { float* p = (float*)d_ws + off; off += nf; return p; };
  float* gl1   = alloc((size_t)NT * 128); // layer2: gl2 (NT*64) + gr2 (NT*64)
  float* gr1   = alloc((size_t)NT * 128);
  float* agg1  = alloc((size_t)NT * 128); // reused as agg2
  float* o1    = alloc((size_t)NT * 128); // reused as node_emb (NT*64)
  float* x1    = alloc((size_t)NT * 128);
  float* logD  = alloc((size_t)(E_DAG + NT) * 2);
  float* exD   = alloc((size_t)(E_DAG + NT) * 2);
  float* mD    = alloc((size_t)NT * 2);
  float* dnD   = alloc((size_t)NT * 2);
  float* glh   = alloc((size_t)NV * 128);
  float* grh   = alloc((size_t)NV * 128);
  float* geh   = alloc((size_t)(E_V2V + NV) * 128);
  float* gefl  = alloc(128);
  float* logH  = alloc((size_t)(E_V2V + NV) * 2);
  float* exH   = alloc((size_t)(E_V2V + NV) * 2);
  float* mH    = alloc((size_t)NV * 2);
  float* dnH   = alloc((size_t)NV * 2);
  float* aggH  = alloc((size_t)NV * 128);
  float* veh1a = alloc((size_t)NV * 128);
  float* veh1b = alloc((size_t)NV * 128);
  float* rsu1  = alloc((size_t)NR * 128);
  float* veh2a = alloc((size_t)NV * 64);
  float* veh2b = alloc((size_t)NV * 64);
  float* rsu2  = alloc((size_t)NR * 64);
  float* vemb  = alloc((size_t)NV * 64);
  float* remb  = alloc((size_t)NR * 64);
  float* eamn  = alloc(16);
  float* gemb  = alloc((size_t)Bc * 64);
  float* vmean = alloc((size_t)Bc * 64);

  // ---- DAG encoder ---------------------------------------------------------
  run_gatv2(st, task_x, NT, 16, task_x, NT, 16, dag_s, dag_d, E_DAG, true,
            nullptr, nullptr, dag1, 2, 64, true, o1,
            gl1, gr1, nullptr, nullptr, logD, exD, mD, dnD, agg1);
  layernorm_k<<<cdiv(NT, 256), 256, 0, st>>>(o1, ln1_g, ln1_b, x1, NT, 128, 1);
  float* gl2 = gl1;
  float* gr2 = gl1 + (size_t)NT * 64;
  float* node_emb = o1;
  run_gatv2(st, x1, NT, 128, x1, NT, 128, dag_s, dag_d, E_DAG, true,
            nullptr, nullptr, dag2, 1, 64, false, node_emb,
            gl2, gr2, nullptr, nullptr, logD, exD, mD, dnD, agg1);

  // ---- hetero topology encoder --------------------------------------------
  col_mean2_k<<<1, 64, 0, st>>>(ea_v2v, eamn, E_V2V);
  run_gatv2(st, veh_x, NV, 12, veh_x, NV, 12, v2v_s, v2v_d, E_V2V, true,
            ea_v2v, eamn, h1_v2v, 2, 64, true, veh1a,
            glh, grh, geh, gefl, logH, exH, mH, dnH, aggH);
  run_gatv2(st, rsu_x, NR, 8, veh_x, NV, 12, i2v_s, i2v_d, E_VI, false,
            ea_i2v, nullptr, h1_i2v, 2, 64, true, veh1b,
            glh, grh, geh, gefl, logH, exH, mH, dnH, aggH);
  run_gatv2(st, veh_x, NV, 12, rsu_x, NR, 8, v2i_s, v2i_d, E_VI, false,
            ea_v2i, nullptr, h1_v2i, 2, 64, true, rsu1,
            glh, grh, geh, gefl, logH, exH, mH, dnH, aggH);
  ewise_k<<<cdiv(NV * 128, 256), 256, 0, st>>>(veh1a, veh1b, veh1a, NV * 128, 1);
  ewise_k<<<cdiv(NR * 128, 256), 256, 0, st>>>(rsu1, nullptr, rsu1, NR * 128, 1);
  run_gatv2(st, veh1a, NV, 128, veh1a, NV, 128, v2v_s, v2v_d, E_V2V, true,
            ea_v2v, eamn, h2_v2v, 1, 64, false, veh2a,
            glh, grh, geh, gefl, logH, exH, mH, dnH, aggH);
  run_gatv2(st, rsu1, NR, 128, veh1a, NV, 128, i2v_s, i2v_d, E_VI, false,
            ea_i2v, nullptr, h2_i2v, 1, 64, false, veh2b,
            glh, grh, geh, gefl, logH, exH, mH, dnH, aggH);
  run_gatv2(st, veh1a, NV, 128, rsu1, NR, 128, v2i_s, v2i_d, E_VI, false,
            ea_v2i, nullptr, h2_v2i, 1, 64, false, rsu2,
            glh, grh, geh, gefl, logH, exH, mH, dnH, aggH);
  ewise_k<<<cdiv(NV * 64, 256), 256, 0, st>>>(veh2a, veh2b, veh2a, NV * 64, 0);
  layernorm_k<<<cdiv(NV, 256), 256, 0, st>>>(veh2a, lnv_g, lnv_b, vemb, NV, 64, 0);
  layernorm_k<<<cdiv(NR, 256), 256, 0, st>>>(rsu2, lnr_g, lnr_b, remb, NR, 64, 0);

  // ---- pooling, scores, power ---------------------------------------------
  mean_pool_k<<<cdiv(Bc * 64, 256), 256, 0, st>>>(node_emb, gemb, Bc, Tc, 64);
  mean_pool_k<<<cdiv(Bc * 64, 256), 256, 0, st>>>(vemb, vmean, Bc, Vc, 64);

  float* logits = (float*)d_out;                       // [B, T*40]
  float* power  = (float*)d_out + (size_t)Bc * Tc * KTGT;
  scores_wmma_k<<<dim3(cdiv(KTGT, 16), Tc / 16, Bc), 32, 0, st>>>(
      node_emb, vemb, remb, mask, logits);
  power_k<<<Bc, 32, 0, st>>>(gemb, vmean, pW1, pb1, pW2, pb2, power);
}
